// DynamicGRU_70824010711544
// MI455X (gfx1250) — matrix operand order repaired and checked by reading, not compiled, not run on hardware
//
#include <hip/hip_runtime.h>
#include <hip/hip_bf16.h>
#include <stdint.h>
#include <stddef.h>

// Problem dims
#define B_    32
#define T_    512
#define D_    1024
#define N_    1024
#define KTOT  2048          // D + N
#define NKT   64            // KTOT / 32 (bf16 WMMA K-tile = 32)
#define NTG   128           // (2N)/16 gate column tiles
#define NTC   64            // N/16 candidate column tiles

// Persistent kernel shape
#define NBLOCKS 32
#define THREADS 128         // 4 waves (wave32)
#define WPB     4

typedef __attribute__((ext_vector_type(16))) __bf16       v16bf;
typedef __attribute__((ext_vector_type(8)))  float        v8f;
typedef __attribute__((ext_vector_type(8)))  unsigned int v8u;

union Frag {
    v16bf          bf;
    v8u            u8;
    unsigned short s[16];
};

__device__ __forceinline__ unsigned short f2bf(float f) {
    unsigned u = __float_as_uint(f);
    u += 0x7FFFu + ((u >> 16) & 1u);      // round-to-nearest-even
    return (unsigned short)(u >> 16);
}

// Agent-scope f32 load/store: cross-workgroup visible (bypass stale WGP$)
__device__ __forceinline__ float ldv(const float* p) {
    return __hip_atomic_load((float*)p, __ATOMIC_RELAXED, __HIP_MEMORY_SCOPE_AGENT);
}
__device__ __forceinline__ void stv(float* p, float v) {
    __hip_atomic_store(p, v, __ATOMIC_RELAXED, __HIP_MEMORY_SCOPE_AGENT);
}

__device__ __forceinline__ float sigmoidf_(float x) {
    return 1.0f / (1.0f + __expf(-x));
}

__device__ __forceinline__ v8f wmma_bf16(v16bf a, v16bf b, v8f c) {
    return __builtin_amdgcn_wmma_f32_16x16x32_bf16(false, a, false, b, (short)0, c, false, false);
}

// A-fragment (16x32 bf16) inverse index: where element (row m, col k-within-1024) lives.
// A layout per lane (l = fh*16 + (m&15)), element e: K = fh*16 + e
__device__ __forceinline__ size_t frag_off(int row, int n) {
    int kt  = n >> 5;
    int kin = n & 31;
    int fh  = (kin >> 4) & 1;
    int e   = kin & 15;
    int mt  = row >> 4;
    int lfr = fh * 16 + (row & 15);
    return (((size_t)kt * 2 + mt) * 32 + lfr) * 16 + e;
}

// ---- grid-wide generation barrier (bar[0]=arrive count, bar[1]=generation) ----
__device__ __forceinline__ void grid_sync(unsigned* bar, unsigned* phase) {
    __threadfence();
    __syncthreads();
    if (threadIdx.x == 0) {
        const unsigned target = ++(*phase);
        unsigned arrived = __hip_atomic_fetch_add(&bar[0], 1u, __ATOMIC_ACQ_REL,
                                                  __HIP_MEMORY_SCOPE_AGENT);
        if (arrived == (unsigned)(gridDim.x - 1)) {
            __hip_atomic_store(&bar[0], 0u, __ATOMIC_RELAXED, __HIP_MEMORY_SCOPE_AGENT);
            __hip_atomic_store(&bar[1], target, __ATOMIC_RELEASE, __HIP_MEMORY_SCOPE_AGENT);
        } else {
            while (__hip_atomic_load(&bar[1], __ATOMIC_ACQUIRE,
                                     __HIP_MEMORY_SCOPE_AGENT) < target) {
                __builtin_amdgcn_s_sleep(2);
            }
        }
    }
    __syncthreads();
}

// ---- setup: zero barrier, h(f32) = h0, h fragment (bf16) = h0 ----
__global__ void gru_init(const float* __restrict__ h0, float* __restrict__ h,
                         unsigned short* __restrict__ hf, unsigned* __restrict__ bar) {
    int i = blockIdx.x * blockDim.x + threadIdx.x;
    if (i < 2) bar[i] = 0u;
    if (i < B_ * N_) {
        int b = i >> 10, n = i & 1023;
        float v = h0[i];
        h[i] = v;
        hf[frag_off(b, n)] = f2bf(v);
    }
}

// ---- setup: Wg [2048,2048] f32 -> bf16 [nt][kt][lane][e] (B-fragment layout) ----
// B layout: element e of lane l: K = kt*32 + (l>>4)*16 + e, N = nt*16 + (l&15)
__global__ void conv_wg(const float* __restrict__ Wg, unsigned short* __restrict__ dst) {
    int i = blockIdx.x * blockDim.x + threadIdx.x;   // coalesced read: i = k*2048 + n
    if (i >= KTOT * 2 * N_) return;
    int k = i >> 11, n = i & 2047;
    int kt = k >> 5, kin = k & 31;
    int hf = kin >> 4, e = kin & 15;
    int nt = n >> 4;
    int lane = (hf << 4) | (n & 15);
    size_t o = (((size_t)nt * NKT + kt) * 32 + lane) * 16 + e;
    dst[o] = f2bf(Wg[i]);
}

// ---- setup: Wc [2048,1024] f32 -> bf16 [nt][kt][lane][e] ----
__global__ void conv_wc(const float* __restrict__ Wc, unsigned short* __restrict__ dst) {
    int i = blockIdx.x * blockDim.x + threadIdx.x;   // i = k*1024 + n
    if (i >= KTOT * N_) return;
    int k = i >> 10, n = i & 1023;
    int kt = k >> 5, kin = k & 31;
    int hf = kin >> 4, e = kin & 15;
    int nt = n >> 4;
    int lane = (hf << 4) | (n & 15);
    size_t o = (((size_t)nt * NKT + kt) * 32 + lane) * 16 + e;
    dst[o] = f2bf(Wc[i]);
}

// ---- setup: x [B,T,D] f32 -> bf16 A-fragments [t][kt][mt][lane][e] ----
__global__ void conv_x(const float* __restrict__ x, unsigned short* __restrict__ dst) {
    int i = blockIdx.x * blockDim.x + threadIdx.x;   // i = (b*T + t)*D + k
    if (i >= B_ * T_ * D_) return;
    int b = i >> 19;
    int t = (i >> 10) & (T_ - 1);
    int k = i & 1023;
    size_t o = (size_t)t * (32 * 2 * 512) + frag_off(b, k);
    dst[o] = f2bf(x[i]);
}

// ---- persistent GRU recurrence ----
__global__ __launch_bounds__(THREADS, 1)
void gru_persistent(const float* __restrict__ bg,
                    const float* __restrict__ bc,
                    const unsigned short* __restrict__ wg,
                    const unsigned short* __restrict__ wc,
                    const unsigned short* __restrict__ xf,
                    float* h, float* u,
                    unsigned short* hf, unsigned short* rhf,
                    unsigned* bar,
                    float* __restrict__ out) {
    const int lane = threadIdx.x & 31;
    const int wid  = blockIdx.x * WPB + (threadIdx.x >> 5);   // 0..127
    const int half = lane >> 4;
    const int l15  = lane & 15;
    const int lo16 = lane * 16;
    unsigned phase = 0;

    for (int t = 0; t < T_; ++t) {
        const unsigned short* xfb = xf + (size_t)t * (32 * 2 * 512);

        // ===== phase 1: gates = sigmoid([x_t, h] @ Wg + bg) =====
        {
            const int nt = wid;                                   // 0..127
            const unsigned short* wb = wg + (size_t)nt * (NKT * 512);
            v8f a0a = {0,0,0,0,0,0,0,0}, a0b = {0,0,0,0,0,0,0,0};
            v8f a1a = {0,0,0,0,0,0,0,0}, a1b = {0,0,0,0,0,0,0,0};
            // K tiles 0..31: A from x fragments (read-only, plain loads)
            for (int kt = 0; kt < 32; kt += 2) {
                Frag b0, b1, f00, f10, f01, f11;
                b0.u8  = *(const v8u*)(wb + (size_t)kt * 512 + lo16);
                b1.u8  = *(const v8u*)(wb + (size_t)(kt + 1) * 512 + lo16);
                f00.u8 = *(const v8u*)(xfb + ((size_t)kt * 2 + 0) * 512 + lo16);
                f10.u8 = *(const v8u*)(xfb + ((size_t)kt * 2 + 1) * 512 + lo16);
                f01.u8 = *(const v8u*)(xfb + ((size_t)(kt + 1) * 2 + 0) * 512 + lo16);
                f11.u8 = *(const v8u*)(xfb + ((size_t)(kt + 1) * 2 + 1) * 512 + lo16);
                a0a = wmma_bf16(f00.bf, b0.bf, a0a);
                a1a = wmma_bf16(f10.bf, b0.bf, a1a);
                a0b = wmma_bf16(f01.bf, b1.bf, a0b);
                a1b = wmma_bf16(f11.bf, b1.bf, a1b);
            }
            // K tiles 32..63: A from h fragments (cross-block, volatile loads)
            for (int q = 0; q < 32; q += 2) {
                Frag b0, b1, f00, f10, f01, f11;
                b0.u8  = *(const v8u*)(wb + (size_t)(32 + q) * 512 + lo16);
                b1.u8  = *(const v8u*)(wb + (size_t)(33 + q) * 512 + lo16);
                f00.u8 = *(volatile const v8u*)(hf + ((size_t)q * 2 + 0) * 512 + lo16);
                f10.u8 = *(volatile const v8u*)(hf + ((size_t)q * 2 + 1) * 512 + lo16);
                f01.u8 = *(volatile const v8u*)(hf + ((size_t)(q + 1) * 2 + 0) * 512 + lo16);
                f11.u8 = *(volatile const v8u*)(hf + ((size_t)(q + 1) * 2 + 1) * 512 + lo16);
                a0a = wmma_bf16(f00.bf, b0.bf, a0a);
                a1a = wmma_bf16(f10.bf, b0.bf, a1a);
                a0b = wmma_bf16(f01.bf, b1.bf, a0b);
                a1b = wmma_bf16(f11.bf, b1.bf, a1b);
            }
            v8f acc0 = a0a + a0b;
            v8f acc1 = a1a + a1b;

            const int n    = nt * 16 + l15;       // column in [0, 2048)
            const float bs = bg[n];
#pragma unroll
            for (int i = 0; i < 8; ++i) {
                int r0 = i + 8 * half;            // batch rows (C/D layout)
                int r1 = 16 + r0;
                float g0 = sigmoidf_(acc0[i] + bs);
                float g1 = sigmoidf_(acc1[i] + bs);
                if (n < N_) {                     // r gate -> write (r*h) bf16 fragments
                    float rv0 = g0 * ldv(h + (size_t)r0 * N_ + n);
                    float rv1 = g1 * ldv(h + (size_t)r1 * N_ + n);
                    *(volatile unsigned short*)(rhf + frag_off(r0, n)) = f2bf(rv0);
                    *(volatile unsigned short*)(rhf + frag_off(r1, n)) = f2bf(rv1);
                } else {                          // u gate
                    stv(u + (size_t)r0 * N_ + (n - N_), g0);
                    stv(u + (size_t)r1 * N_ + (n - N_), g1);
                }
            }
        }
        grid_sync(bar, &phase);

        // ===== phase 2: c = tanh([x_t, r*h] @ Wc + bc); h = u*h + (1-u)*c =====
        if (wid < NTC) {
            const int nt = wid;                                   // 0..63
            const unsigned short* wb = wc + (size_t)nt * (NKT * 512);
            v8f a0a = {0,0,0,0,0,0,0,0}, a0b = {0,0,0,0,0,0,0,0};
            v8f a1a = {0,0,0,0,0,0,0,0}, a1b = {0,0,0,0,0,0,0,0};
            for (int kt = 0; kt < 32; kt += 2) {
                Frag b0, b1, f00, f10, f01, f11;
                b0.u8  = *(const v8u*)(wb + (size_t)kt * 512 + lo16);
                b1.u8  = *(const v8u*)(wb + (size_t)(kt + 1) * 512 + lo16);
                f00.u8 = *(const v8u*)(xfb + ((size_t)kt * 2 + 0) * 512 + lo16);
                f10.u8 = *(const v8u*)(xfb + ((size_t)kt * 2 + 1) * 512 + lo16);
                f01.u8 = *(const v8u*)(xfb + ((size_t)(kt + 1) * 2 + 0) * 512 + lo16);
                f11.u8 = *(const v8u*)(xfb + ((size_t)(kt + 1) * 2 + 1) * 512 + lo16);
                a0a = wmma_bf16(f00.bf, b0.bf, a0a);
                a1a = wmma_bf16(f10.bf, b0.bf, a1a);
                a0b = wmma_bf16(f01.bf, b1.bf, a0b);
                a1b = wmma_bf16(f11.bf, b1.bf, a1b);
            }
            for (int q = 0; q < 32; q += 2) {
                Frag b0, b1, f00, f10, f01, f11;
                b0.u8  = *(const v8u*)(wb + (size_t)(32 + q) * 512 + lo16);
                b1.u8  = *(const v8u*)(wb + (size_t)(33 + q) * 512 + lo16);
                f00.u8 = *(volatile const v8u*)(rhf + ((size_t)q * 2 + 0) * 512 + lo16);
                f10.u8 = *(volatile const v8u*)(rhf + ((size_t)q * 2 + 1) * 512 + lo16);
                f01.u8 = *(volatile const v8u*)(rhf + ((size_t)(q + 1) * 2 + 0) * 512 + lo16);
                f11.u8 = *(volatile const v8u*)(rhf + ((size_t)(q + 1) * 2 + 1) * 512 + lo16);
                a0a = wmma_bf16(f00.bf, b0.bf, a0a);
                a1a = wmma_bf16(f10.bf, b0.bf, a1a);
                a0b = wmma_bf16(f01.bf, b1.bf, a0b);
                a1b = wmma_bf16(f11.bf, b1.bf, a1b);
            }
            v8f acc0 = a0a + a0b;
            v8f acc1 = a1a + a1b;

            const int n    = nt * 16 + l15;       // column in [0, 1024)
            const float bs = bc[n];
#pragma unroll
            for (int i = 0; i < 8; ++i) {
                int r0 = i + 8 * half;
                int r1 = 16 + r0;
                float c0 = tanhf(acc0[i] + bs);
                float c1 = tanhf(acc1[i] + bs);
                float u0 = ldv(u + (size_t)r0 * N_ + n);
                float u1 = ldv(u + (size_t)r1 * N_ + n);
                float h0 = ldv(h + (size_t)r0 * N_ + n);
                float h1 = ldv(h + (size_t)r1 * N_ + n);
                float hn0 = u0 * h0 + (1.0f - u0) * c0;
                float hn1 = u1 * h1 + (1.0f - u1) * c1;
                stv(h + (size_t)r0 * N_ + n, hn0);
                stv(h + (size_t)r1 * N_ + n, hn1);
                *(volatile unsigned short*)(hf + frag_off(r0, n)) = f2bf(hn0);
                *(volatile unsigned short*)(hf + frag_off(r1, n)) = f2bf(hn1);
                out[((size_t)r0 * T_ + t) * N_ + n] = hn0;   // activations row = b*T + t
                out[((size_t)r1 * T_ + t) * N_ + n] = hn1;
            }
        }
        grid_sync(bar, &phase);
    }
}

extern "C" void kernel_launch(void* const* d_in, const int* in_sizes, int n_in,
                              void* d_out, int out_size, void* d_ws, size_t ws_size,
                              hipStream_t stream) {
    (void)in_sizes; (void)n_in; (void)out_size; (void)ws_size;
    const float* x  = (const float*)d_in[0];   // [B,T,D]
    const float* h0 = (const float*)d_in[1];   // [B,N]
    const float* Wg = (const float*)d_in[2];   // [D+N, 2N]
    const float* bg = (const float*)d_in[3];   // [2N]
    const float* Wc = (const float*)d_in[4];   // [D+N, N]
    const float* bc = (const float*)d_in[5];   // [N]
    float* out = (float*)d_out;                // [B*T, N]

    // workspace layout
    char* ws = (char*)d_ws;
    size_t o = 0;
    unsigned*       bar = (unsigned*)(ws + o);        o += 256;
    unsigned short* wgb = (unsigned short*)(ws + o);  o += (size_t)NKT * NTG * 512 * 2; // 8 MB
    unsigned short* wcb = (unsigned short*)(ws + o);  o += (size_t)NKT * NTC * 512 * 2; // 4 MB
    unsigned short* xfb = (unsigned short*)(ws + o);  o += (size_t)T_ * 32 * 2 * 512 * 2; // 32 MB
    float*          h   = (float*)(ws + o);           o += (size_t)B_ * N_ * 4;  // 128 KB
    float*          u   = (float*)(ws + o);           o += (size_t)B_ * N_ * 4;  // 128 KB
    unsigned short* hfr = (unsigned short*)(ws + o);  o += 32 * 2 * 512 * 2;     // 64 KB
    unsigned short* rhf = (unsigned short*)(ws + o);  o += 32 * 2 * 512 * 2;     // 64 KB

    gru_init<<<(B_ * N_ + 255) / 256, 256, 0, stream>>>(h0, h, hfr, bar);
    conv_wg<<<(KTOT * 2 * N_ + 255) / 256, 256, 0, stream>>>(Wg, wgb);
    conv_wc<<<(KTOT * N_ + 255) / 256, 256, 0, stream>>>(Wc, wcb);
    conv_x<<<(B_ * T_ * D_ + 255) / 256, 256, 0, stream>>>(x, xfb);
    gru_persistent<<<NBLOCKS, THREADS, 0, stream>>>(bg, bc, wgb, wcb, xfb,
                                                    h, u, hfr, rhf, bar, out);
}
